// QuerySAT_6176162971908
// MI455X (gfx1250) — matrix operand order repaired and checked by reading, not compiled, not run on hardware
//
#include <hip/hip_runtime.h>
#include <stdint.h>

#define F 128

typedef __attribute__((ext_vector_type(16))) _Float16 v16h;
typedef __attribute__((ext_vector_type(8)))  float    v8f;

union Frag16 { v16h v; _Float16 h[16]; uint32_t u[8]; };

__device__ __forceinline__ float softplus_f(float x) {
    // numerically stable softplus
    return fmaxf(x, 0.0f) + log1pf(__expf(-fabsf(x)));
}

// ---------------------------------------------------------------------------
// Pack weight W[K,128] (f32 row-major) into WMMA B-fragment layout:
// P[kt][nt][reg j][lane] = half2( W[k0][n], W[k0+1][n] )
//   lane: n = nt*16 + (lane&15); half = lane>>4
//   k0   = kt*32 + half*16 + j*2   (B 32x16: lanes 0-15 hold K=0..15, 16-31 K=16..31)
// ---------------------------------------------------------------------------
__global__ void pack_w_kernel(const float* __restrict__ W, uint32_t* __restrict__ P, int K) {
    int idx = blockIdx.x * blockDim.x + threadIdx.x;
    int total = (K >> 5) * 2048;
    if (idx >= total) return;
    int lane = idx & 31;
    int j    = (idx >> 5) & 7;
    int nt   = (idx >> 8) & 7;
    int kt   = idx >> 11;
    int half = lane >> 4;
    int n    = nt * 16 + (lane & 15);
    int k0   = kt * 32 + half * 16 + j * 2;
    union { _Float16 h[2]; uint32_t u; } cv;
    cv.h[0] = (_Float16)W[(size_t)k0 * F + n];
    cv.h[1] = (_Float16)W[(size_t)(k0 + 1) * F + n];
    P[idx] = cv.u;
}

// ---------------------------------------------------------------------------
// Fused GEMM + bias + activation:  Out[M,128] = act(A[M,K] @ Wpk + bias)
//  - packed f16 weights staged through LDS once per block (shared by 8 waves)
//  - B fragments double-buffered out of LDS across the unrolled nt loop
//  - K and ACT compile-time: k-loop fully unrolled (K/32 * 8 WMMAs)
//  - In-place (Out==A, K==128) is safe: each wave reads only its own 16 rows
//    across the whole k-loop before writing them.
// ACT: 0=none, 1=relu, 2=sigmoid
// ---------------------------------------------------------------------------
template <int K, int ACT>
__global__ __launch_bounds__(256) void gemm_mlp_kernel(
    const float* __restrict__ A, int M,
    const uint32_t* __restrict__ Wpk, const float* __restrict__ bias,
    float* __restrict__ Out)
{
    constexpr int KT = K >> 5;              // k-tiles of 32
    __shared__ uint32_t smem[K * 64];       // full packed weight (32/64 KB)

    // cooperative staging: global -> LDS with b128 copies
    {
        const uint4* __restrict__ src = (const uint4*)Wpk;
        uint4* dst = (uint4*)smem;
        constexpr int NV4 = (K * 64) >> 2;  // 2048 or 4096
#pragma unroll
        for (int i = 0; i < NV4 / 256; ++i)
            dst[i * 256 + threadIdx.x] = src[i * 256 + threadIdx.x];
    }
    __syncthreads();

    const int wave = threadIdx.x >> 5;
    const int lane = threadIdx.x & 31;
    const int row0 = (blockIdx.x * 8 + wave) * 16;

    if (row0 < M) {
        const int half = lane >> 4;
        const int ncol = lane & 15;
        int mload = row0 + ncol;
        if (mload >= M) mload = M - 1;       // clamp so EXEC stays all-1s
        const float* __restrict__ arow = A + (size_t)mload * K;

        float bv[8];
#pragma unroll
        for (int nt = 0; nt < 8; ++nt) bv[nt] = bias[nt * 16 + ncol];

        v8f zero = {0.f, 0.f, 0.f, 0.f, 0.f, 0.f, 0.f, 0.f};
        v8f acc[8];
#pragma unroll
        for (int i = 0; i < 8; ++i) acc[i] = zero;

#pragma unroll
        for (int kt = 0; kt < KT; ++kt) {
            // A fragment: 16x32 f16, converted from f32 in-register.
            // reg j holds K pair k = half*16 + j*2
            Frag16 a;
#pragma unroll
            for (int j = 0; j < 8; ++j) {
                int k = kt * 32 + (half * 16 + j * 2);
                float2 f = *(const float2*)(arow + k);
                a.h[2 * j]     = (_Float16)f.x;
                a.h[2 * j + 1] = (_Float16)f.y;
            }

            const uint32_t* __restrict__ wp = smem + kt * 2048 + lane;
            Frag16 bb[2];
#pragma unroll
            for (int j = 0; j < 8; ++j) bb[0].u[j] = wp[j * 32];

#pragma unroll
            for (int nt = 0; nt < 8; ++nt) {
                if (nt < 7) {
#pragma unroll
                    for (int j = 0; j < 8; ++j)
                        bb[(nt + 1) & 1].u[j] = wp[(nt + 1) * 256 + j * 32];
                }
                acc[nt] = __builtin_amdgcn_wmma_f32_16x16x32_f16(
                    false, a.v, false, bb[nt & 1].v, (short)0, acc[nt], false, false);
            }
        }

        // Epilogue. D layout: reg r -> row = r + 8*half; col = ncol (+ 16*nt)
#pragma unroll
        for (int nt = 0; nt < 8; ++nt) {
#pragma unroll
            for (int r = 0; r < 8; ++r) {
                int mrow = row0 + half * 8 + r;
                if (mrow < M) {
                    float v = acc[nt][r] + bv[nt];
                    if (ACT == 1)      v = fmaxf(v, 0.0f);
                    else if (ACT == 2) v = 1.0f / (1.0f + __expf(-v));
                    Out[(size_t)mrow * F + nt * 16 + ncol] = v;
                }
            }
        }
    }
}

// variables[v,0:256] = concat(literals[v], literals[v+V])
__global__ void build_vars_kernel(const float* __restrict__ L, float* __restrict__ VA, int V) {
    int idx = blockIdx.x * blockDim.x + threadIdx.x;
    if (idx >= V * 2 * F) return;
    int v = idx >> 8;
    int k = idx & 255;
    VA[idx] = (k < F) ? L[(size_t)v * F + k] : L[(size_t)(v + V) * F + (k - F)];
}

// clauses_val[c,f] = exp(-sum_j softplus(sign_j * logits[var_j, f]))
__global__ void clause_val_kernel(const float* __restrict__ logits,
                                  const int* __restrict__ cvar, const int* __restrict__ cneg,
                                  float* __restrict__ cv, int C) {
    int idx = blockIdx.x * blockDim.x + threadIdx.x;
    if (idx >= C * F) return;
    int c = idx >> 7;
    int f = idx & (F - 1);
    float s = 0.0f;
#pragma unroll
    for (int j = 0; j < 3; ++j) {
        int var = cvar[c * 3 + j];
        float sign = cneg[c * 3 + j] ? -1.0f : 1.0f;
        s += softplus_f(sign * logits[(size_t)var * F + f]);
    }
    cv[(size_t)c * F + f] = __expf(-s);
}

__global__ void zero_kernel(float* __restrict__ p, int n) {
    int idx = blockIdx.x * blockDim.x + threadIdx.x;
    if (idx < n) p[idx] = 0.0f;
}

// scatter-add clause messages into literal rows (f32 global atomics, L2-resident)
__global__ void scatter_kernel(const float* __restrict__ closs,
                               const int* __restrict__ cvar, const int* __restrict__ cneg,
                               float* __restrict__ lloss, int C, int V) {
    int idx = blockIdx.x * blockDim.x + threadIdx.x;
    if (idx >= C * 3 * F) return;
    int f = idx & (F - 1);
    int e = idx >> 7;            // c*3 + j
    int c = e / 3;
    int j = e - c * 3;
    int var = cvar[c * 3 + j];
    int lit = cneg[c * 3 + j] ? (var + V) : var;
    unsafeAtomicAdd(&lloss[(size_t)lit * F + f], closs[(size_t)c * F + f]);
}

// unit = layernorm(flip(concat(literals, literals_loss))) ; one wave per row
__global__ __launch_bounds__(256) void unit_ln_kernel(
    const float* __restrict__ L, const float* __restrict__ LL,
    const float* __restrict__ g, const float* __restrict__ b,
    float* __restrict__ U, int V)
{
    int wid  = (blockIdx.x * blockDim.x + threadIdx.x) >> 5;
    int lane = threadIdx.x & 31;
    int nlits = 2 * V;
    if (wid >= nlits) return;
    int src = (wid < V) ? (wid + V) : (wid - V);
    const float* __restrict__ lrow = L  + (size_t)src * F;
    const float* __restrict__ srow = LL + (size_t)src * F;

    float x[8], s = 0.0f, ss = 0.0f;
#pragma unroll
    for (int i = 0; i < 8; ++i) {
        int k = lane + i * 32;
        float v = (k < F) ? lrow[k] : srow[k - F];
        x[i] = v; s += v; ss += v * v;
    }
#pragma unroll
    for (int off = 16; off; off >>= 1) {
        s  += __shfl_xor(s,  off, 32);
        ss += __shfl_xor(ss, off, 32);
    }
    float mean = s * (1.0f / 256.0f);
    float var  = ss * (1.0f / 256.0f) - mean * mean;
    float inv  = rsqrtf(var + 1e-3f);
    float* __restrict__ urow = U + (size_t)wid * 2 * F;
#pragma unroll
    for (int i = 0; i < 8; ++i) {
        int k = lane + i * 32;
        urow[k] = (x[i] - mean) * inv * g[k] + b[k];
    }
}

// literals = (1-g)*literals + g*new
__global__ void combine_kernel(float* __restrict__ L, const float* __restrict__ G,
                               const float* __restrict__ NW, int n) {
    int idx = blockIdx.x * blockDim.x + threadIdx.x;
    if (idx >= n) return;
    float gg = G[idx];
    L[idx] = (1.0f - gg) * L[idx] + gg * NW[idx];
}

// final vote layer: out[v] = dot(H[v,:], w[:,0]) + b  ; one wave per row
__global__ __launch_bounds__(256) void vote_out_kernel(
    const float* __restrict__ H, const float* __restrict__ w,
    const float* __restrict__ b, float* __restrict__ out, int V)
{
    int wid  = (blockIdx.x * blockDim.x + threadIdx.x) >> 5;
    int lane = threadIdx.x & 31;
    if (wid >= V) return;
    const float* __restrict__ hr = H + (size_t)wid * F;
    float s = 0.0f;
#pragma unroll
    for (int i = 0; i < 4; ++i) s += hr[lane + i * 32] * w[lane + i * 32];
#pragma unroll
    for (int off = 16; off; off >>= 1) s += __shfl_xor(s, off, 32);
    if (lane == 0) out[wid] = s + b[0];
}

// ---------------------------------------------------------------------------
extern "C" void kernel_launch(void* const* d_in, const int* in_sizes, int n_in,
                              void* d_out, int out_size, void* d_ws, size_t ws_size,
                              hipStream_t stream)
{
    const int* cvar  = (const int*)d_in[0];
    const int* cneg  = (const int*)d_in[1];
    const float* lit0 = (const float*)d_in[2];
    const int C     = in_sizes[0] / 3;
    const int nlits = in_sizes[2] / F;
    const int V     = nlits / 2;
    const int maxM  = (C > nlits) ? C : nlits;

    // groups: 0=query, 1=query_inter, 2=forget, 3=update, 4=vote
    const float* W[5][3]; const float* B[5][3];
    int base = 3;
    for (int g = 0; g < 5; ++g) {
        for (int i = 0; i < 3; ++i) {
            W[g][i] = (const float*)d_in[base + 2 * i];
            B[g][i] = (const float*)d_in[base + 2 * i + 1];
        }
        base += 6;
    }
    const float* ln_g = (const float*)d_in[33];
    const float* ln_b = (const float*)d_in[34];

    char* ws = (char*)d_ws;
    size_t off = 0;
    auto carve = [&](size_t bytes) -> char* {
        char* p = ws + off;
        off += (bytes + 255) & ~(size_t)255;
        return p;
    };

    // packed f16 weights for all 128-wide layers (vote w2 stays f32)
    static const int KDIM[5][3] = { {2*F,F,F}, {F,F,F}, {2*F,F,F}, {2*F,F,F}, {2*F,F,0} };
    uint32_t* PK[5][3] = {};
    for (int g = 0; g < 5; ++g)
        for (int i = 0; i < 3; ++i)
            if (KDIM[g][i]) PK[g][i] = (uint32_t*)carve((size_t)KDIM[g][i] * 64 * sizeof(uint32_t));

    float* L    = (float*)carve((size_t)nlits * F * 4);       // persistent literals
    float* T256 = (float*)carve((size_t)nlits * 2 * F * 4);   // variables / unit
    float* H    = (float*)carve((size_t)maxM * F * 4);        // MLP hidden
    float* CB   = (float*)carve((size_t)maxM * F * 4);        // clauses_val/loss, later NW
    float* LG   = (float*)carve((size_t)V * F * 4);           // logits
    float* LL   = (float*)carve((size_t)nlits * F * 4);       // literals_loss, later G
    float* G    = LL;                                         // reuse after unit built
    float* NW   = CB;                                         // reuse after scatter
    (void)ws_size; (void)n_in; (void)out_size;

    // one-time weight packing
    for (int g = 0; g < 5; ++g)
        for (int i = 0; i < 3; ++i)
            if (KDIM[g][i]) {
                int total = KDIM[g][i] * 64;
                pack_w_kernel<<<(total + 255) / 256, 256, 0, stream>>>(W[g][i], PK[g][i], KDIM[g][i]);
            }

    hipMemcpyAsync(L, lit0, (size_t)nlits * F * 4, hipMemcpyDeviceToDevice, stream);

    // K=256 layers are always relu (layer 0 of query/forget/update/vote)
    auto gemm = [&](const float* A, int M, int K, const uint32_t* pk, const float* bias,
                    float* Out, int act) {
        dim3 grid((M + 127) / 128);
        if (K == 2 * F)
            gemm_mlp_kernel<2 * F, 1><<<grid, 256, 0, stream>>>(A, M, pk, bias, Out);
        else if (act == 0)
            gemm_mlp_kernel<F, 0><<<grid, 256, 0, stream>>>(A, M, pk, bias, Out);
        else if (act == 1)
            gemm_mlp_kernel<F, 1><<<grid, 256, 0, stream>>>(A, M, pk, bias, Out);
        else
            gemm_mlp_kernel<F, 2><<<grid, 256, 0, stream>>>(A, M, pk, bias, Out);
    };

    for (int r = 0; r < 16; ++r) {
        // query MLP on variables
        build_vars_kernel<<<(V * 2 * F + 255) / 256, 256, 0, stream>>>(L, T256, V);
        gemm(T256, V, 2 * F, PK[0][0], B[0][0], H, 1);
        gemm(H,    V, F,     PK[0][1], B[0][1], H, 1);
        gemm(H,    V, F,     PK[0][2], B[0][2], LG, 0);
        // clause reduction + query_inter MLP
        clause_val_kernel<<<(C * F + 255) / 256, 256, 0, stream>>>(LG, cvar, cneg, CB, C);
        gemm(CB, C, F, PK[1][0], B[1][0], H, 1);
        gemm(H,  C, F, PK[1][1], B[1][1], H, 1);
        gemm(H,  C, F, PK[1][2], B[1][2], CB, 0);
        // scatter into literal rows
        zero_kernel<<<(nlits * F + 255) / 256, 256, 0, stream>>>(LL, nlits * F);
        scatter_kernel<<<(C * 3 * F + 255) / 256, 256, 0, stream>>>(CB, cvar, cneg, LL, C, V);
        // flip + layernorm
        unit_ln_kernel<<<(nlits + 7) / 8, 256, 0, stream>>>(L, LL, ln_g, ln_b, T256, V);
        // forget MLP (sigmoid) -> G (reuses LL)
        gemm(T256, nlits, 2 * F, PK[2][0], B[2][0], H, 1);
        gemm(H,    nlits, F,     PK[2][1], B[2][1], H, 1);
        gemm(H,    nlits, F,     PK[2][2], B[2][2], G, 2);
        // update MLP (relu) -> NW (reuses CB)
        gemm(T256, nlits, 2 * F, PK[3][0], B[3][0], H, 1);
        gemm(H,    nlits, F,     PK[3][1], B[3][1], H, 1);
        gemm(H,    nlits, F,     PK[3][2], B[3][2], NW, 1);
        // gated combine
        combine_kernel<<<(nlits * F + 255) / 256, 256, 0, stream>>>(L, G, NW, nlits * F);
    }

    // vote head
    build_vars_kernel<<<(V * 2 * F + 255) / 256, 256, 0, stream>>>(L, T256, V);
    gemm(T256, V, 2 * F, PK[4][0], B[4][0], H, 1);
    gemm(H,    V, F,     PK[4][1], B[4][1], H, 1);
    vote_out_kernel<<<(V + 7) / 8, 256, 0, stream>>>(H, W[4][2], B[4][2], (float*)d_out, V);
}
